// TemporalGNN_21354577395749
// MI455X (gfx1250) — compile-verified, matrix operand order checked by reading, not run on hardware
//
#include <hip/hip_runtime.h>
#include <hip/hip_fp16.h>

#define TT   12
#define FF   8
#define HIDN 32
#define OUTC 36

typedef __attribute__((ext_vector_type(16))) _Float16 v16h;
typedef __attribute__((ext_vector_type(8)))  float    v8f;

// Branch-free nonlinearities. gfx1250 has a hardware v_tanh_f32 trans op.
__device__ __forceinline__ float fast_tanh(float x) {
#if __has_builtin(__builtin_amdgcn_tanhf)
  return __builtin_amdgcn_tanhf(x);
#else
  // tanh(x) = 1 - 2/(e^{2x}+1); branch-free, saturates correctly in fp32
  float e = __expf(2.0f * x);
  return 1.0f - 2.0f / (e + 1.0f);
#endif
}
__device__ __forceinline__ float fast_sigmoid(float x) {
  // sigmoid(x) = 0.5 * tanh(x/2) + 0.5  (single trans op)
  return fmaf(0.5f, fast_tanh(0.5f * x), 0.5f);
}

// ---------------- kernel 1: in-degree count (dst side, u32 atomics) ----------------
__global__ __launch_bounds__(256) void k_degree(const int* __restrict__ ei,
                                                unsigned* __restrict__ deg, int E) {
  int i = blockIdx.x * 256 + threadIdx.x;
  if (i < E) atomicAdd(&deg[ei[E + i]], 1u);
}

// ---------------- kernel 2: dinv = rsqrt(deg + 1 self-loop), in place ----------------
__global__ __launch_bounds__(256) void k_dinv(float* __restrict__ dinv_io, int n) {
  int i = blockIdx.x * 256 + threadIdx.x;
  if (i < n) {
    unsigned d = reinterpret_cast<unsigned*>(dinv_io)[i];
    dinv_io[i] = rsqrtf((float)(d + 1u));
  }
}

// ---------------- kernel 3: prep — softmax probs, folded weights in WMMA B layout ----
// 16-bit A/B fragment layout (wave32): element e, lane l ->
//   k = (e<8 ? e : e+8) + (l>=16 ? 8 : 0),  row/col = l & 15
__global__ __launch_bounds__(256) void k_prep(
    const float* __restrict__ Wcz, const float* __restrict__ bcz,
    const float* __restrict__ Wch, const float* __restrict__ bch,
    const float* __restrict__ Wlz, const float* __restrict__ blz,
    const float* __restrict__ Wlh, const float* __restrict__ blh,
    const float* __restrict__ att, const float* __restrict__ Wout,
    _Float16* __restrict__ BfZ, _Float16* __restrict__ BfH, _Float16* __restrict__ BfO,
    float* __restrict__ bz_eff, float* __restrict__ bh_eff, float* __restrict__ probs) {
  int tid = threadIdx.x;
  // softmax(attention) over T=12 (redundant small serial loops)
  if (tid < TT) {
    float m = att[0];
    for (int i = 1; i < TT; ++i) m = fmaxf(m, att[i]);
    float s = 0.f;
    for (int i = 0; i < TT; ++i) s += __expf(att[i] - m);
    probs[tid] = __expf(att[tid] - m) / s;
  }
  // gate B fragments: Weff = Wc(8x32) @ Wl[:32](32x32), K padded 8->32 with zeros
  for (int idx = tid; idx < 2048; idx += 256) {
    int e = idx & 15, lane = (idx >> 4) & 31, tile = (idx >> 9) & 1, gate = idx >> 10;
    int k = (e < 8 ? e : e + 8) + ((lane >= 16) ? 8 : 0);
    int col = tile * 16 + (lane & 15);
    float v = 0.f;
    if (lane < 16 && k < FF) {
      const float* Wc = gate ? Wch : Wcz;
      const float* Wl = gate ? Wlh : Wlz;
      for (int q = 0; q < HIDN; ++q) v += Wc[k * HIDN + q] * Wl[q * HIDN + col];
    }
    (gate ? BfH : BfZ)[(tile * 32 + lane) * 16 + e] = (_Float16)v;
  }
  // effective biases: b_eff = bc @ Wl[:32] + bl
  for (int idx = tid; idx < 64; idx += 256) {
    int gate = idx >> 5, c = idx & 31;
    const float* bc = gate ? bch : bcz;
    const float* Wl = gate ? Wlh : Wlz;
    const float* bl = gate ? blh : blz;
    float v = bl[c];
    for (int q = 0; q < HIDN; ++q) v += bc[q] * Wl[q * HIDN + c];
    (gate ? bh_eff : bz_eff)[c] = v;
  }
  // output-head B fragments: Wout(32x36) padded to 32x48, full K=32 used
  for (int idx = tid; idx < 1536; idx += 256) {
    int e = idx & 15, lane = (idx >> 4) & 31, tile = idx >> 9;
    int k = (e < 8 ? e : e + 8) + ((lane >= 16) ? 8 : 0);
    int col = tile * 16 + (lane & 15);
    float v = (col < OUTC) ? Wout[k * OUTC + col] : 0.f;
    BfO[(tile * 32 + lane) * 16 + e] = (_Float16)v;
  }
}

// ---------------- kernel 4: edge scatter — AXf[dst, t, f] += norm * x[src, f, t] -----
__global__ __launch_bounds__(256) void k_edge(const float* __restrict__ x,
                                              const int* __restrict__ ei,
                                              const float* __restrict__ dinv,
                                              float* __restrict__ AXf, int E) {
  unsigned gtid = blockIdx.x * 256 + threadIdx.x;
  int e = (int)(gtid >> 5);
  int lane = threadIdx.x & 31;
  if (e >= E) return;
  int src = ei[e];
  int dst = ei[E + e];
  float nrm = dinv[src] * dinv[dst];
  const float* xs = x + (size_t)src * 96;
  float* ax = AXf + (size_t)dst * 96;
  if (e + 4 < E) {  // speculative prefetch of an upcoming source row
    __builtin_prefetch(x + (size_t)ei[e + 4] * 96, 0, 0);
  }
#pragma unroll
  for (int j = 0; j < 3; ++j) {
    int k = lane + j * 32;       // x channel index = f*12 + t (coalesced read)
    int f = k / TT, t = k % TT;
    unsafeAtomicAdd(&ax[t * FF + f], nrm * xs[k]);
  }
}

// ---------------- kernel 5: add self-loop term, convert to f16 (t-major) -------------
__global__ __launch_bounds__(256) void k_convert(const float* __restrict__ AXf,
                                                 const float* __restrict__ x,
                                                 const float* __restrict__ dinv,
                                                 _Float16* __restrict__ AXh, int n) {
  int idx = blockIdx.x * 256 + threadIdx.x;
  if (idx >= n * 96) return;
  int node = idx / 96, i = idx % 96;
  int t = i >> 3, f = i & 7;
  float dd = dinv[node];
  float self = dd * dd * x[(size_t)node * 96 + f * TT + t];
  AXh[idx] = (_Float16)(AXf[idx] + self);
}

// ---------------- kernel 6: per-node WMMA kernel (16 nodes per wave) -----------------
__global__ __launch_bounds__(256) void k_node(const _Float16* __restrict__ AXh,
                                              const _Float16* __restrict__ BfZ,
                                              const _Float16* __restrict__ BfH,
                                              const _Float16* __restrict__ BfO,
                                              const float* __restrict__ bz_eff,
                                              const float* __restrict__ bh_eff,
                                              const float* __restrict__ probs,
                                              const float* __restrict__ bout,
                                              float* __restrict__ out, int n) {
  __shared__ float smem[8][16 * 32];   // per-wave 16x32 tile for C->A relayout
  int wave = threadIdx.x >> 5;
  int lane = threadIdx.x & 31;
  int m0 = (blockIdx.x * 8 + wave) * 16;  // first node of this wave's tile
  int nn = lane & 15;
  int Mofs = (lane >= 16) ? 8 : 0;

  const v16h* BZ = (const v16h*)BfZ;
  const v16h* BH = (const v16h*)BfH;
  const v16h* BO = (const v16h*)BfO;
  v16h bz0 = BZ[lane], bz1 = BZ[32 + lane];
  v16h bh0 = BH[lane], bh1 = BH[32 + lane];
  v16h bo0 = BO[lane], bo1 = BO[32 + lane], bo2 = BO[64 + lane];

  float bzlo = bz_eff[nn], bzhi = bz_eff[16 + nn];
  float bhlo = bh_eff[nn], bhhi = bh_eff[16 + nn];
  float pr[TT];
#pragma unroll
  for (int t = 0; t < TT; ++t) pr[t] = probs[t];

  v8f czero = {};
  v8f acc0 = {}, acc1 = {};
  for (int t = 0; t < TT; ++t) {
    v16h a = {};
    if (lane < 16 && (m0 + lane) < n) {
      const _Float16* p = AXh + (size_t)(m0 + lane) * 96 + t * FF;
#pragma unroll
      for (int f = 0; f < FF; ++f) a[f] = p[f];   // K=0..7; rest zero-padded
    }
    v8f z0 = __builtin_amdgcn_wmma_f32_16x16x32_f16(false, a, false, bz0, (short)0, czero, false, false);
    v8f z1 = __builtin_amdgcn_wmma_f32_16x16x32_f16(false, a, false, bz1, (short)0, czero, false, false);
    v8f h0 = __builtin_amdgcn_wmma_f32_16x16x32_f16(false, a, false, bh0, (short)0, czero, false, false);
    v8f h1 = __builtin_amdgcn_wmma_f32_16x16x32_f16(false, a, false, bh1, (short)0, czero, false, false);
    float p = pr[t];
#pragma unroll
    for (int r = 0; r < 8; ++r) {
      float zz = fast_sigmoid(z0[r] + bzlo);
      float hh = fast_tanh(h0[r] + bhlo);
      acc0[r] += p * (1.f - zz) * hh;
      zz = fast_sigmoid(z1[r] + bzhi);
      hh = fast_tanh(h1[r] + bhhi);
      acc1[r] += p * (1.f - zz) * hh;
    }
  }

  // relu(H_acc): write in C/D layout, re-read in A layout through LDS
  __syncthreads();
  float* L = &smem[wave][0];
#pragma unroll
  for (int r = 0; r < 8; ++r) {
    L[(r + Mofs) * 32 + nn]      = fmaxf(acc0[r], 0.f);
    L[(r + Mofs) * 32 + 16 + nn] = fmaxf(acc1[r], 0.f);
  }
  __syncthreads();
  v16h ao;
#pragma unroll
  for (int e = 0; e < 16; ++e) {
    int k = (e < 8 ? e : e + 8) + ((lane >= 16) ? 8 : 0);   // K = HID index, fully used
    ao[e] = (_Float16)L[nn * 32 + k];
  }
  v8f d0 = __builtin_amdgcn_wmma_f32_16x16x32_f16(false, ao, false, bo0, (short)0, czero, false, false);
  v8f d1 = __builtin_amdgcn_wmma_f32_16x16x32_f16(false, ao, false, bo1, (short)0, czero, false, false);
  v8f d2 = __builtin_amdgcn_wmma_f32_16x16x32_f16(false, ao, false, bo2, (short)0, czero, false, false);
  v8f dt[3] = {d0, d1, d2};
#pragma unroll
  for (int j = 0; j < 3; ++j) {
    int col = j * 16 + nn;
    if (col < OUTC) {
      float bb = bout[col];
#pragma unroll
      for (int r = 0; r < 8; ++r) {
        int node = m0 + r + Mofs;
        if (node < n) out[(size_t)node * OUTC + col] = dt[j][r] + bb;
      }
    }
  }
}

extern "C" void kernel_launch(void* const* d_in, const int* in_sizes, int n_in,
                              void* d_out, int out_size, void* d_ws, size_t ws_size,
                              hipStream_t stream) {
  const float* x    = (const float*)d_in[0];
  const int*   ei   = (const int*)d_in[1];
  const float* Wcz  = (const float*)d_in[2];
  const float* bcz  = (const float*)d_in[3];
  // d_in[4], d_in[5] (Wcr/bcr) and d_in[10], d_in[11] (Wlr/blr) are mathematically
  // dead: H0==0 every period so the R gate never reaches the output.
  const float* Wch  = (const float*)d_in[6];
  const float* bch  = (const float*)d_in[7];
  const float* Wlz  = (const float*)d_in[8];
  const float* blz  = (const float*)d_in[9];
  const float* Wlh  = (const float*)d_in[12];
  const float* blh  = (const float*)d_in[13];
  const float* att  = (const float*)d_in[14];
  const float* Wout = (const float*)d_in[15];
  const float* bout = (const float*)d_in[16];
  float* out = (float*)d_out;

  int Nn = in_sizes[0] / (FF * TT);   // 50000
  int E  = in_sizes[1] / 2;           // 1600000

  auto al = [](size_t v) { return (v + 255) & ~(size_t)255; };
  char* ws = (char*)d_ws;
  size_t o = 0;
  float*    dinv = (float*)(ws + o);     o += al((size_t)Nn * 4);
  float*    AXf  = (float*)(ws + o);     o += al((size_t)Nn * 96 * 4);
  _Float16* AXh  = (_Float16*)(ws + o);  o += al((size_t)Nn * 96 * 2);
  _Float16* BfZ  = (_Float16*)(ws + o);  o += al(2 * 32 * 16 * 2);
  _Float16* BfH  = (_Float16*)(ws + o);  o += al(2 * 32 * 16 * 2);
  _Float16* BfO  = (_Float16*)(ws + o);  o += al(3 * 32 * 16 * 2);
  float* bz_eff  = (float*)(ws + o);     o += al(HIDN * 4);
  float* bh_eff  = (float*)(ws + o);     o += al(HIDN * 4);
  float* probs   = (float*)(ws + o);     o += al(TT * 4);
  (void)ws_size; (void)n_in; (void)out_size;

  hipMemsetAsync(dinv, 0, (size_t)Nn * 4, stream);
  hipMemsetAsync(AXf, 0, (size_t)Nn * 96 * 4, stream);

  k_degree<<<(E + 255) / 256, 256, 0, stream>>>(ei, (unsigned*)dinv, E);
  k_dinv<<<(Nn + 255) / 256, 256, 0, stream>>>(dinv, Nn);
  k_prep<<<1, 256, 0, stream>>>(Wcz, bcz, Wch, bch, Wlz, blz, Wlh, blh, att, Wout,
                                BfZ, BfH, BfO, bz_eff, bh_eff, probs);
  unsigned eblocks = (unsigned)((((long long)E * 32) + 255) / 256);
  k_edge<<<eblocks, 256, 0, stream>>>(x, ei, dinv, AXf, E);
  k_convert<<<((Nn * 96) + 255) / 256, 256, 0, stream>>>(AXf, x, dinv, AXh, Nn);
  int waves = (Nn + 15) / 16;
  k_node<<<(waves + 7) / 8, 256, 0, stream>>>(AXh, BfZ, BfH, BfO, bz_eff, bh_eff,
                                              probs, bout, out, Nn);
}
